// DeepRC_14345190769306
// MI455X (gfx1250) — matrix operand order repaired and checked by reading, not compiled, not run on hardware
//
#include <hip/hip_runtime.h>
#include <math.h>

// ---------------------------------------------------------------------------
// DeepRC attention-MIL forward on gfx1250 (MI455X).
// Conv-as-GEMM via sliding-window K' = s*24+c trick; all matrix math in
// v_wmma_f32_16x16x32_f16 with f32 accumulation; SELU/max in VALU.
// ---------------------------------------------------------------------------

typedef _Float16 h8   __attribute__((ext_vector_type(8)));
typedef _Float16 v16h __attribute__((ext_vector_type(16)));
typedef float    v8f  __attribute__((ext_vector_type(8)));

__device__ __forceinline__ v16h ld_frag_split(const _Float16* lo, const _Float16* hi) {
  h8 a = *(const h8*)lo;   // one ds_load_b128 (16B aligned)
  h8 b = *(const h8*)hi;   // one ds_load_b128
  v16h r;
#pragma unroll
  for (int i = 0; i < 8; ++i) { r[i] = a[i]; r[i + 8] = b[i]; }
  return r;
}
__device__ __forceinline__ v16h ld_frag16(const _Float16* p) {
  return ld_frag_split(p, p + 8);   // contiguous 16 halfwords
}
__device__ __forceinline__ float selu1(float x) {
  const float sc = 1.0507009873554805f, al = 1.6732632423543772f;
  return sc * (x > 0.f ? x : al * (__expf(x) - 1.f));
}

// ===========================================================================
// Kernel 1: embed (conv+SELU+maxpool) + attention MLP logits.
// One wave (32 threads) per block, 16 instances per block.
// LDS: conv-B [32][224] f16 (14336B) + W1/W2 [32][32] f16 (4096B)
//      + x-tile 16*[776] f16 (24832B)  = 43264B
// ===========================================================================
__global__ __launch_bounds__(32)
void k1_embed_att(const float* __restrict__ x,
                  const float* __restrict__ convw,   // (32,23,9)
                  const float* __restrict__ convb,   // (32)
                  const float* __restrict__ w1, const float* __restrict__ b1,
                  const float* __restrict__ w2, const float* __restrict__ b2,
                  const float* __restrict__ w3, const float* __restrict__ b3,
                  float* __restrict__ emb_out,       // (N,32)
                  float* __restrict__ att_out)       // (N)
{
  __shared__ alignas(16) _Float16 sB [32 * 224];  // conv B: [n][K'], K'=s*24+c
  __shared__ alignas(16) _Float16 sW1[32 * 32];   // [u][k]
  __shared__ alignas(16) _Float16 sW2[32 * 32];
  __shared__ alignas(16) _Float16 sT [16 * 776];  // x tiles, 776 hw/instance

  const int  l     = threadIdx.x;     // lane 0..31
  const int  h     = l >> 4;          // lane half
  const int  cn    = l & 15;
  const long inst0 = (long)blockIdx.x * 16;

  // ---- stage conv weights as B matrix (zero pad c==23 and s>=9) ----
#pragma unroll 1
  for (int n = 0; n < 32; ++n) {
#pragma unroll
    for (int kk = 0; kk < 7; ++kk) {
      int K = kk * 32 + l;
      int s = K / 24, c = K - s * 24;
      float v = (s < 9 && c < 23) ? convw[n * 207 + c * 9 + s] : 0.f;
      sB[n * 224 + K] = (_Float16)v;
    }
  }
  // ---- attention weights: row-major [u][k] is exactly B layout ----
#pragma unroll 1
  for (int u = 0; u < 32; ++u) {
    sW1[u * 32 + l] = (_Float16)w1[u * 32 + l];
    sW2[u * 32 + l] = (_Float16)w2[u * 32 + l];
  }
  // ---- stage 16 instances: lane l copies row l (23 ch) of each instance ----
#pragma unroll 1
  for (int m = 0; m < 16; ++m) {
    const float* src = x + (inst0 + m) * 736 + l * 23;
    _Float16*    dst = sT + m * 776 + l * 24;
#pragma unroll
    for (int t = 0; t < 23; ++t) dst[t] = (_Float16)src[t];
    dst[23] = (_Float16)0.f;                       // pad channel
  }
  if (l < 8) {                                     // zero tail 768..775
#pragma unroll 1
    for (int m = 0; m < 16; ++m) sT[m * 776 + 768 + l] = (_Float16)0.f;
  }
  asm volatile("s_wait_dscnt 0x0" ::: "memory");   // LDS stores visible

  // ---- conv B fragments resident in VGPRs (7 K-chunks x 2 N-tiles) ----
  v16h Bc[7][2];
#pragma unroll
  for (int kc = 0; kc < 7; ++kc)
#pragma unroll
    for (int nh = 0; nh < 2; ++nh)
      Bc[kc][nh] = ld_frag16(sB + (cn + nh * 16) * 224 + kc * 32 + h * 16);

  const float bias0 = convb[cn], bias1 = convb[16 + cn];
  v8f e0, e1;
#pragma unroll
  for (int r = 0; r < 8; ++r) { e0[r] = -3.4e38f; e1[r] = -3.4e38f; }

  const _Float16* tb = sT + cn * 776;  // A-fragment row base (m = lane&15)
#pragma unroll 1
  for (int p = 0; p < 24; ++p) {
    v8f a0 = {0,0,0,0,0,0,0,0}, a1 = {0,0,0,0,0,0,0,0};
#pragma unroll
    for (int kc = 0; kc < 7; ++kc) {
      const _Float16* pa = tb + p * 24 + kc * 32 + h * 16;  // 16B aligned
      v16h A = ld_frag16(pa);
      a0 = __builtin_amdgcn_wmma_f32_16x16x32_f16(false, A, false, Bc[kc][0],
                                                  (short)0, a0, false, false);
      a1 = __builtin_amdgcn_wmma_f32_16x16x32_f16(false, A, false, Bc[kc][1],
                                                  (short)0, a1, false, false);
    }
#pragma unroll
    for (int r = 0; r < 8; ++r) {
      e0[r] = fmaxf(e0[r], selu1(a0[r] + bias0));
      e1[r] = fmaxf(e1[r], selu1(a1[r] + bias1));
    }
  }

  // ---- emb: C-layout -> LDS bounce (f16 [16][32]) + global f32 ----
  _Float16* sE = sT;                   // x tile is dead now; reuse
  _Float16* sH = sT + 512;
  float*    sF = (float*)(sT + 1024);
#pragma unroll
  for (int r = 0; r < 8; ++r) {
    int m = r + 8 * h;
    sE[m * 32 + cn]      = (_Float16)e0[r];
    sE[m * 32 + 16 + cn] = (_Float16)e1[r];
    emb_out[(inst0 + m) * 32 + cn]      = e0[r];
    emb_out[(inst0 + m) * 32 + 16 + cn] = e1[r];
  }
  asm volatile("s_wait_dscnt 0x0" ::: "memory");

  // ---- attention layer 1: h1 = selu(emb @ w1.T + b1), 16x32x32 GEMM ----
  v16h W1a = ld_frag16(sW1 + cn * 32 + h * 16);
  v16h W1b = ld_frag16(sW1 + (cn + 16) * 32 + h * 16);
  {
    const _Float16* pe = sE + cn * 32 + h * 16;
    v16h Ae = ld_frag16(pe);
    v8f g0 = {0,0,0,0,0,0,0,0}, g1 = {0,0,0,0,0,0,0,0};
    g0 = __builtin_amdgcn_wmma_f32_16x16x32_f16(false, Ae, false, W1a, (short)0, g0, false, false);
    g1 = __builtin_amdgcn_wmma_f32_16x16x32_f16(false, Ae, false, W1b, (short)0, g1, false, false);
    float bb0 = b1[cn], bb1 = b1[16 + cn];
#pragma unroll
    for (int r = 0; r < 8; ++r) {
      int m = r + 8 * h;
      sH[m * 32 + cn]      = (_Float16)selu1(g0[r] + bb0);
      sH[m * 32 + 16 + cn] = (_Float16)selu1(g1[r] + bb1);
    }
  }
  asm volatile("s_wait_dscnt 0x0" ::: "memory");

  // ---- attention layer 2: h2 = selu(h1 @ w2.T + b2) -> f32 LDS ----
  v16h W2a = ld_frag16(sW2 + cn * 32 + h * 16);
  v16h W2b = ld_frag16(sW2 + (cn + 16) * 32 + h * 16);
  {
    const _Float16* ph = sH + cn * 32 + h * 16;
    v16h Ah = ld_frag16(ph);
    v8f g0 = {0,0,0,0,0,0,0,0}, g1 = {0,0,0,0,0,0,0,0};
    g0 = __builtin_amdgcn_wmma_f32_16x16x32_f16(false, Ah, false, W2a, (short)0, g0, false, false);
    g1 = __builtin_amdgcn_wmma_f32_16x16x32_f16(false, Ah, false, W2b, (short)0, g1, false, false);
    float bb0 = b2[cn], bb1 = b2[16 + cn];
#pragma unroll
    for (int r = 0; r < 8; ++r) {
      int m = r + 8 * h;
      sF[m * 32 + cn]      = selu1(g0[r] + bb0);
      sF[m * 32 + 16 + cn] = selu1(g1[r] + bb1);
    }
  }
  asm volatile("s_wait_dscnt 0x0" ::: "memory");

  // ---- logits: att[m] = h2[m] . w3 + b3 (lanes 0..15) ----
  if (l < 16) {
    float a = b3[0];
#pragma unroll
    for (int u = 0; u < 32; ++u) a += sF[l * 32 + u] * w3[u];
    att_out[inst0 + l] = a;
  }
}

// ===========================================================================
// Kernel 2a/2b: deterministic per-bag online softmax stats (m, sum exp).
// ===========================================================================
__global__ __launch_bounds__(256)
void k2a_stats_part(const float* __restrict__ att, const int* __restrict__ seg,
                    int n, float* __restrict__ part /* [grid][8][2] */)
{
  __shared__ float sm[256][16];
  float mx[8], su[8];
  int tid = threadIdx.x;
#pragma unroll
  for (int b = 0; b < 8; ++b) { mx[b] = -1e30f; su[b] = 0.f; }
  for (int i = blockIdx.x * 256 + tid; i < n; i += gridDim.x * 256) {
    int b = seg[i]; float a = att[i];
    if (a > mx[b]) { su[b] = su[b] * __expf(mx[b] - a) + 1.f; mx[b] = a; }
    else           { su[b] += __expf(a - mx[b]); }
  }
#pragma unroll
  for (int b = 0; b < 8; ++b) { sm[tid][2 * b] = mx[b]; sm[tid][2 * b + 1] = su[b]; }
  __syncthreads();
  if (tid < 8) {
    float M = -1e30f, S = 0.f;
    for (int t = 0; t < 256; ++t) {
      float m2 = sm[t][2 * tid], s2 = sm[t][2 * tid + 1];
      float Mn = fmaxf(M, m2);
      S = S * __expf(M - Mn) + s2 * __expf(m2 - Mn);
      M = Mn;
    }
    part[blockIdx.x * 16 + tid * 2]     = M;
    part[blockIdx.x * 16 + tid * 2 + 1] = S;
  }
}

__global__ __launch_bounds__(32)
void k2b_stats_final(const float* __restrict__ part, int nparts,
                     float* __restrict__ stats /* m[8], z[8] */)
{
  int tid = threadIdx.x;
  if (tid < 8) {
    float M = -1e30f, S = 0.f;
    for (int p = 0; p < nparts; ++p) {
      float m2 = part[p * 16 + tid * 2], s2 = part[p * 16 + tid * 2 + 1];
      float Mn = fmaxf(M, m2);
      S = S * __expf(M - Mn) + s2 * __expf(m2 - Mn);
      M = Mn;
    }
    stats[tid] = M; stats[8 + tid] = S;
  }
}

// ===========================================================================
// Kernel 3: per-block attention-weighted partial sums of emb (8 bags x 32).
// ===========================================================================
__global__ __launch_bounds__(256)
void k3_pool_part(const float* __restrict__ emb, const float* __restrict__ att,
                  const int* __restrict__ seg, const float* __restrict__ stats,
                  float* __restrict__ part3 /* [grid][8][32] */)
{
  __shared__ float pp[8 * 32];
  int tid = threadIdx.x;
  pp[tid] = 0.f;
  __syncthreads();
  int g = tid >> 5, k = tid & 31;
  long base = (long)blockIdx.x * 256 + g * 32;
  for (int ii = 0; ii < 32; ++ii) {
    long i = base + ii;
    int  b = seg[i];
    float w = __expf(att[i] - stats[b]) / stats[8 + b];
    atomicAdd(&pp[b * 32 + k], emb[i * 32 + k] * w);   // ds_add_f32
  }
  __syncthreads();
  part3[(long)blockIdx.x * 256 + tid] = pp[tid];
}

// ===========================================================================
// Kernel 4: reduce partials, final linear -> (8,1)
// ===========================================================================
__global__ __launch_bounds__(256)
void k4_final(const float* __restrict__ part3, int nblk,
              const float* __restrict__ outw, const float* __restrict__ outb,
              float* __restrict__ out)
{
  __shared__ float pooled[256];
  int tid = threadIdx.x;
  float acc = 0.f;
  for (int b = 0; b < nblk; ++b) acc += part3[(long)b * 256 + tid];
  pooled[tid] = acc;
  __syncthreads();
  if (tid < 8) {
    float o = outb[0];
#pragma unroll
    for (int k = 0; k < 32; ++k) o += pooled[tid * 32 + k] * outw[k];
    out[tid] = o;
  }
}

// ===========================================================================
extern "C" void kernel_launch(void* const* d_in, const int* in_sizes, int n_in,
                              void* d_out, int out_size, void* d_ws, size_t ws_size,
                              hipStream_t stream)
{
  const float* x     = (const float*)d_in[0];
  const int*   seg   = (const int*)  d_in[1];
  const float* convw = (const float*)d_in[2];
  const float* convb = (const float*)d_in[3];
  const float* w1    = (const float*)d_in[4];
  const float* b1    = (const float*)d_in[5];
  const float* w2    = (const float*)d_in[6];
  const float* b2    = (const float*)d_in[7];
  const float* w3    = (const float*)d_in[8];
  const float* b3    = (const float*)d_in[9];
  const float* outw  = (const float*)d_in[10];
  const float* outb  = (const float*)d_in[11];
  float* out = (float*)d_out;

  const int N = in_sizes[1];                  // 131072 instances
  float* ws    = (float*)d_ws;                // needs ~17.8 MB
  float* emb   = ws;                          // N*32
  float* att   = emb + (size_t)N * 32;        // N
  float* part2 = att + N;                     // 64*16
  float* stats = part2 + 64 * 16;             // 16
  float* part3 = stats + 16;                  // N

  k1_embed_att<<<N / 16, 32, 0, stream>>>(x, convw, convb, w1, b1, w2, b2,
                                          w3, b3, emb, att);
  k2a_stats_part<<<64, 256, 0, stream>>>(att, seg, N, part2);
  k2b_stats_final<<<1, 32, 0, stream>>>(part2, 64, stats);
  k3_pool_part<<<N / 256, 256, 0, stream>>>(emb, att, seg, stats, part3);
  k4_final<<<1, 256, 0, stream>>>(part3, N / 256, outw, outb, out);
}
